// CausalSelfAttention_227
// MI455X (gfx1250) — matrix operand order repaired and checked by reading, not compiled, not run on hardware
//
#include <hip/hip_runtime.h>
#include <hip/hip_bf16.h>

typedef _Float16 f16;
typedef __attribute__((ext_vector_type(8)))  _Float16 v8h_t;
typedef __attribute__((ext_vector_type(16))) _Float16 v16h;
typedef __attribute__((ext_vector_type(8)))  float    v8f;

typedef int v4i __attribute__((ext_vector_type(4)));
typedef __attribute__((address_space(1))) v4i av4i_g;   // global int4
typedef __attribute__((address_space(3))) v4i av4i_l;   // LDS int4

#define NB 4
#define NT 2048
#define NC 768
#define NH 12
#define ND 64
#define NM (NB*NT)      // 8192 rows
#define N3 (3*NC)       // 2304

static __device__ __forceinline__ v16h cat8(v8h_t lo, v8h_t hi) {
  return __builtin_shufflevector(lo, hi, 0,1,2,3,4,5,6,7,8,9,10,11,12,13,14,15);
}

static __device__ __forceinline__ v8f wmma_f16(v16h a, v16h b, v8f c) {
  // D = A x B + C, fp32 accumulate  ->  v_wmma_f32_16x16x32_f16
  return __builtin_amdgcn_wmma_f32_16x16x32_f16(false, a, false, b, (short)0, c, false, false);
}

// ---- CDNA5 async global->LDS copy (ASYNCcnt path), with safe fallback ----
#if __has_builtin(__builtin_amdgcn_global_load_async_to_lds_b128)
#define HAS_ASYNC_LDS 1
#else
#define HAS_ASYNC_LDS 0
#endif

static __device__ __forceinline__ void async_copy16(const f16* g, f16* l) {
#if HAS_ASYNC_LDS
  __builtin_amdgcn_global_load_async_to_lds_b128(
      (av4i_g*)(uintptr_t)g, (av4i_l*)(uint32_t)(uintptr_t)l, 0, 0);
#else
  *(v8h_t*)l = *(const v8h_t*)g;
#endif
}

static __device__ __forceinline__ void async_wait0() {
#if HAS_ASYNC_LDS
#if __has_builtin(__builtin_amdgcn_s_wait_asynccnt)
  __builtin_amdgcn_s_wait_asynccnt(0);
#else
  asm volatile("s_wait_asynccnt 0x0" ::: "memory");
#endif
#endif
}

// ---------------------------------------------------------------- converts
__global__ __launch_bounds__(256) void k_f32_to_f16(const float* __restrict__ src,
                                                    f16* __restrict__ dst, int n) {
  int i = blockIdx.x * 256 + threadIdx.x;
  if (i < n) dst[i] = (f16)src[i];
}

// src[rows][cols] f32 -> dst[cols][rows] f16   (K-contiguous B operand layout)
__global__ __launch_bounds__(256) void k_transpose_f16(const float* __restrict__ src,
                                                       f16* __restrict__ dst,
                                                       int rows, int cols) {
  int c = blockIdx.x * 16 + threadIdx.x;
  int r = blockIdx.y * 16 + threadIdx.y;
  if (r < rows && c < cols) dst[(size_t)c * rows + r] = (f16)src[(size_t)r * cols + c];
}

// ---------------------------------------------------------------- QKV GEMM
// xh [8192,768] f16 (row major, K contig), waT [2304,768] f16 ([N][K], K contig)
// writes q,k as [B,H,T,D] f16 and v transposed as [B,H,D,T] f16, bias added.
__global__ __launch_bounds__(128) void k_gemm_qkv(const f16* __restrict__ xh,
                                                  const f16* __restrict__ waT,
                                                  const float* __restrict__ bias,
                                                  f16* __restrict__ qo,
                                                  f16* __restrict__ ko,
                                                  f16* __restrict__ vT) {
  const int lane = threadIdx.x & 31;
  const int wid  = threadIdx.x >> 5;
  const int hi   = lane >> 4;          // half-wave select
  const int ln   = lane & 15;
  const int mtile = blockIdx.x * 4 + wid;   // 0..511 (16 rows each)
  const int ngrp  = blockIdx.y;             // 0..35  (64 cols each)

  const f16* aptr = xh + (size_t)(mtile * 16 + ln) * NC;

  v8f acc[4] = {};
  for (int kk = 0; kk < NC; kk += 32) {
    const f16* ap = aptr + kk + hi * 8;
    v16h a = cat8(*(const v8h_t*)(ap), *(const v8h_t*)(ap + 16));
#pragma unroll
    for (int t = 0; t < 4; ++t) {
      const int n = ngrp * 64 + t * 16 + ln;
      const f16* bp = waT + (size_t)n * NC + kk + hi * 8;
      v16h b = cat8(*(const v8h_t*)(bp), *(const v8h_t*)(bp + 16));
      acc[t] = wmma_f16(a, b, acc[t]);
    }
  }
#pragma unroll
  for (int t = 0; t < 4; ++t) {
    const int n  = ngrp * 64 + t * 16 + ln;     // global column in [0,2304)
    const float bs = bias[n];
    const int sel = n / NC;                     // 0=q 1=k 2=v
    const int cw  = n % NC;
    const int h = cw >> 6, d = cw & 63;
#pragma unroll
    for (int r = 0; r < 8; ++r) {
      const int mrow = mtile * 16 + r + hi * 8;   // global token row
      const int bidx = mrow >> 11;                // / 2048
      const int tt   = mrow & (NT - 1);
      const float val = acc[t][r] + bs;
      if (sel == 0)      qo[((size_t)(bidx * NH + h) * NT + tt) * ND + d] = (f16)val;
      else if (sel == 1) ko[((size_t)(bidx * NH + h) * NT + tt) * ND + d] = (f16)val;
      else               vT[((size_t)(bidx * NH + h) * ND + d) * NT + tt] = (f16)val;
    }
  }
}

// ---------------------------------------------------------------- attention
// 4 waves per block cover 4 consecutive 16-row q tiles of one (b,h).
// K/V tiles staged once per j into LDS via async global->LDS, shared by all waves.
__global__ __launch_bounds__(128) void k_attn(const f16* __restrict__ q,
                                              const f16* __restrict__ k,
                                              const f16* __restrict__ vT,
                                              f16* __restrict__ out) {  // [B,T,C]
  __shared__ __align__(16) f16 ldsK[16 * 64];      // [key][d]
  __shared__ __align__(16) f16 ldsV[64 * 16];      // [d][key]  (from vT)
  __shared__ __align__(16) f16 pbuf[4][16 * 16];   // per-wave P staging

  const int tid  = threadIdx.x;
  const int lane = tid & 31;
  const int wid  = tid >> 5;
  const int hi = lane >> 4, ln = lane & 15;
  const int qi   = blockIdx.x * 4 + wid;   // this wave's q tile (0..127)
  const int jmax = blockIdx.x * 4 + 3;     // block-uniform loop bound
  const int bh = blockIdx.y;               // 0..47
  const int b  = bh / NH, h = bh % NH;

  const f16* qbase = q  + (size_t)bh * NT * ND;
  const f16* kbase = k  + (size_t)bh * NT * ND;
  const f16* vbase = vT + (size_t)bh * ND * NT;   // [D][T]

  // Q A-fragments: rows qi*16+ln, K = head dim (two K=32 chunks)
  const f16* qp = qbase + (size_t)(qi * 16 + ln) * ND;
  const v16h qa0 = cat8(*(const v8h_t*)(qp + 0  + hi * 8), *(const v8h_t*)(qp + 16 + hi * 8));
  const v16h qa1 = cat8(*(const v8h_t*)(qp + 32 + hi * 8), *(const v8h_t*)(qp + 48 + hi * 8));

  // per-thread staging assignments (128 threads, 128 x 16B chunks each buffer)
  const int krow = tid >> 3, kc = tid & 7;   // K tile: 16 rows x 8 chunks
  const int vrow = tid >> 1, vc = tid & 1;   // V tile: 64 rows x 2 chunks

  v8f o0 = {}, o1 = {}, o2 = {}, o3 = {};
  float m[8], l[8];
#pragma unroll
  for (int r = 0; r < 8; ++r) { m[r] = -3.0e38f; l[r] = 0.0f; }

  const float scale = 0.125f;   // 1/sqrt(64)

  for (int j = 0; j <= jmax; ++j) {
    // ---- stage K/V tile j into LDS (all 128 threads) ----
    async_copy16(kbase + (size_t)(j * 16 + krow) * ND + kc * 8, &ldsK[krow * 64 + kc * 8]);
    async_copy16(vbase + (size_t)vrow * NT + j * 16 + vc * 8,   &ldsV[vrow * 16 + vc * 8]);
    async_wait0();
    __syncthreads();

    if (j <= qi) {   // wave-uniform predicate -> EXEC all-ones at WMMAs
      // S = Q K^T : B operand from ldsK[key][d]
      const f16* kl = &ldsK[ln * 64];
      v16h kb0 = cat8(*(const v8h_t*)(kl + 0  + hi * 8), *(const v8h_t*)(kl + 16 + hi * 8));
      v16h kb1 = cat8(*(const v8h_t*)(kl + 32 + hi * 8), *(const v8h_t*)(kl + 48 + hi * 8));
      v8f s = {};
      s = wmma_f16(qa0, kb0, s);
      s = wmma_f16(qa1, kb1, s);

      float pv[8];
#pragma unroll
      for (int r = 0; r < 8; ++r) {
        float sv = s[r] * scale;
        if (j == qi && ln > (r + hi * 8)) sv = -3.0e38f;   // causal mask, diagonal tile
        pv[r] = sv;
      }
      // online softmax: row stats across the 16-lane half-wave holding each row
#pragma unroll
      for (int r = 0; r < 8; ++r) {
        float rm = pv[r];
        rm = fmaxf(rm, __shfl_xor(rm, 1, 32));
        rm = fmaxf(rm, __shfl_xor(rm, 2, 32));
        rm = fmaxf(rm, __shfl_xor(rm, 4, 32));
        rm = fmaxf(rm, __shfl_xor(rm, 8, 32));
        const float nm = fmaxf(m[r], rm);
        const float p  = __expf(pv[r] - nm);
        float rs = p;
        rs += __shfl_xor(rs, 1, 32);
        rs += __shfl_xor(rs, 2, 32);
        rs += __shfl_xor(rs, 4, 32);
        rs += __shfl_xor(rs, 8, 32);
        const float corr = __expf(m[r] - nm);
        m[r] = nm;
        l[r] = l[r] * corr + rs;
        o0[r] *= corr; o1[r] *= corr; o2[r] *= corr; o3[r] *= corr;
        pv[r] = p;
      }
      // C-layout -> A-layout conversion of P via per-wave LDS buffer.
      // Same-wave DS ops are in-order (DScnt), no barrier needed.
      f16* pb = pbuf[wid];
#pragma unroll
      for (int r = 0; r < 8; ++r) pb[(r + hi * 8) * 16 + ln] = (f16)pv[r];
      v8h_t z8 = {};
      v16h pa = hi == 0 ? cat8(*(const v8h_t*)(&pb[ln * 16]), *(const v8h_t*)(&pb[ln * 16 + 8])) : cat8(z8, z8);  // K 16..31 = 0

      // O += P @ V : B operand from ldsV[d][key]; K>=16 rows zero
      v16h vz = {};
      v16h vb0 = vz, vb1 = vz, vb2 = vz, vb3 = vz;
      if (hi == 0) {
        const f16* vp0 = &ldsV[(0  + ln) * 16];
        const f16* vp1 = &ldsV[(16 + ln) * 16];
        const f16* vp2 = &ldsV[(32 + ln) * 16];
        const f16* vp3 = &ldsV[(48 + ln) * 16];
        vb0 = cat8(*(const v8h_t*)(vp0), *(const v8h_t*)(vp0 + 8));
        vb1 = cat8(*(const v8h_t*)(vp1), *(const v8h_t*)(vp1 + 8));
        vb2 = cat8(*(const v8h_t*)(vp2), *(const v8h_t*)(vp2 + 8));
        vb3 = cat8(*(const v8h_t*)(vp3), *(const v8h_t*)(vp3 + 8));
      }
      o0 = wmma_f16(pa, vb0, o0);
      o1 = wmma_f16(pa, vb1, o1);
      o2 = wmma_f16(pa, vb2, o2);
      o3 = wmma_f16(pa, vb3, o3);
    }
    __syncthreads();   // protect shared K/V before next iteration overwrites
  }

  // normalize and write merged-head output [B,T,C] f16
#pragma unroll
  for (int r = 0; r < 8; ++r) {
    const int trow = qi * 16 + r + hi * 8;
    const float inv = 1.0f / l[r];
    f16* op = out + ((size_t)b * NT + trow) * NC + h * ND;
    op[0  + ln] = (f16)(o0[r] * inv);
    op[16 + ln] = (f16)(o1[r] * inv);
    op[32 + ln] = (f16)(o2[r] * inv);
    op[48 + ln] = (f16)(o3[r] * inv);
  }
}

// ---------------------------------------------------------------- proj GEMM
__global__ __launch_bounds__(128) void k_gemm_proj(const f16* __restrict__ ah,
                                                   const f16* __restrict__ wpT,
                                                   const float* __restrict__ bias,
                                                   float* __restrict__ out) {
  const int lane = threadIdx.x & 31;
  const int wid  = threadIdx.x >> 5;
  const int hi   = lane >> 4;
  const int ln   = lane & 15;
  const int mtile = blockIdx.x * 4 + wid;   // 0..511
  const int ngrp  = blockIdx.y;             // 0..11

  const f16* aptr = ah + (size_t)(mtile * 16 + ln) * NC;

  v8f acc[4] = {};
  for (int kk = 0; kk < NC; kk += 32) {
    const f16* ap = aptr + kk + hi * 8;
    v16h a = cat8(*(const v8h_t*)(ap), *(const v8h_t*)(ap + 16));
#pragma unroll
    for (int t = 0; t < 4; ++t) {
      const int n = ngrp * 64 + t * 16 + ln;
      const f16* bp = wpT + (size_t)n * NC + kk + hi * 8;
      v16h b = cat8(*(const v8h_t*)(bp), *(const v8h_t*)(bp + 16));
      acc[t] = wmma_f16(a, b, acc[t]);
    }
  }
#pragma unroll
  for (int t = 0; t < 4; ++t) {
    const int n = ngrp * 64 + t * 16 + ln;
    const float bs = bias[n];
#pragma unroll
    for (int r = 0; r < 8; ++r) {
      const int mrow = mtile * 16 + r + hi * 8;
      out[(size_t)mrow * NC + n] = acc[t][r] + bs;
    }
  }
}

// ---------------------------------------------------------------- launch
extern "C" void kernel_launch(void* const* d_in, const int* in_sizes, int n_in,
                              void* d_out, int out_size, void* d_ws, size_t ws_size,
                              hipStream_t stream) {
  const float* x      = (const float*)d_in[0];
  const float* W_attn = (const float*)d_in[1];
  const float* b_attn = (const float*)d_in[2];
  const float* W_proj = (const float*)d_in[3];
  const float* b_proj = (const float*)d_in[4];
  float* out = (float*)d_out;

  char* ws = (char*)d_ws;
  const size_t SZ_X   = (size_t)NM * NC * 2;   // 12.6 MB each
  f16* xh    = (f16*)(ws);                               // [8192,768]
  f16* waT   = (f16*)(ws + SZ_X);                        // [2304,768]
  f16* wpT   = (f16*)(ws + SZ_X + (size_t)N3 * NC * 2);  // [768,768]
  char* p    = ws + SZ_X + (size_t)N3 * NC * 2 + (size_t)NC * NC * 2;
  f16* qf    = (f16*)(p);                 // [B,H,T,D]
  f16* kf    = (f16*)(p + SZ_X);          // [B,H,T,D]
  f16* vT    = (f16*)(p + 2 * SZ_X);      // [B,H,D,T]
  f16* attnh = (f16*)(p + 3 * SZ_X);      // [B,T,C]

  // 1) x -> f16
  k_f32_to_f16<<<(NM * NC + 255) / 256, 256, 0, stream>>>(x, xh, NM * NC);
  // 2) W transposes to [N][K] f16
  k_transpose_f16<<<dim3(N3 / 16, NC / 16), dim3(16, 16), 0, stream>>>(W_attn, waT, NC, N3);
  k_transpose_f16<<<dim3(NC / 16, NC / 16), dim3(16, 16), 0, stream>>>(W_proj, wpT, NC, NC);
  // 3) QKV GEMM (+bias, head-major scatter, V transposed)
  k_gemm_qkv<<<dim3(NM / 64, N3 / 64), 128, 0, stream>>>(xh, waT, b_attn, qf, kf, vT);
  // 4) causal flash attention (4 q-tiles per block, shared async K/V staging)
  k_attn<<<dim3(NT / 64, NB * NH), 128, 0, stream>>>(qf, kf, vT, attnh);
  // 5) output projection (+bias) -> f32
  k_gemm_proj<<<dim3(NM / 64, NC / 64), 128, 0, stream>>>(attnh, wpT, b_proj, out);
}
